// GLBLAttentionPathway_65515431133543
// MI455X (gfx1250) — matrix operand order repaired and checked by reading, not compiled, not run on hardware
//
#include <hip/hip_runtime.h>
#include <hip/hip_bf16.h>
#include <stdint.h>

// ---------------------------------------------------------------------------
// Problem constants
// ---------------------------------------------------------------------------
#define BB   8
#define SS   1024
#define DD   512
#define TD   1536          // 3*D
#define PP   16
#define KK   8
#define HH   8
#define CTXH 2
#define RH1_ 128
#define RH2_ 64
#define MM   (BB*SS)       // 8192 tokens

typedef __attribute__((ext_vector_type(16))) _Float16     v16h;
typedef __attribute__((ext_vector_type(8)))  float        v8f;
typedef __attribute__((ext_vector_type(4)))  unsigned int v4u;
typedef __attribute__((ext_vector_type(8)))  int          v8i;
typedef __attribute__((ext_vector_type(4)))  int          v4i;

union FragU { v16h v; uint32_t u[8]; };
union PackU { _Float16 h[2]; uint32_t u; };

// ---------------------------------------------------------------------------
// Wave32 row reductions over 16-lane halves (matches WMMA C-matrix layout)
// ---------------------------------------------------------------------------
__device__ __forceinline__ float rmax16(float x) {
  x = fmaxf(x, __shfl_xor(x, 1, 32));
  x = fmaxf(x, __shfl_xor(x, 2, 32));
  x = fmaxf(x, __shfl_xor(x, 4, 32));
  x = fmaxf(x, __shfl_xor(x, 8, 32));
  return x;
}
__device__ __forceinline__ float rsum16(float x) {
  x += __shfl_xor(x, 1, 32);
  x += __shfl_xor(x, 2, 32);
  x += __shfl_xor(x, 4, 32);
  x += __shfl_xor(x, 8, 32);
  return x;
}

// ---------------------------------------------------------------------------
// Tensor Data Mover: 2D tile load global -> LDS (f16 elements).
// Builds the D# descriptor per CDNA5 ISA and issues
// tensor_load_to_lds (TENSORcnt-tracked).  width/rows in elements,
// stride in elements (row pitch of the global tensor).
// ---------------------------------------------------------------------------
__device__ __forceinline__ void tdm_load_2d_f16(uint32_t lds_addr, const void* gptr,
                                                uint32_t width, uint32_t rows,
                                                uint32_t stride) {
  const uint64_t ga = (uint64_t)(uintptr_t)gptr;
  v4u g0;
  g0.x = 1u;                                                 // count=1, user mode
  g0.y = lds_addr;                                           // lds_addr[31:0]
  g0.z = (uint32_t)ga;                                       // global_addr[31:0]
  g0.w = (uint32_t)((ga >> 32) & 0x01FFFFFFu) | (2u << 30);  // addr[56:32] | type=2
  v8i g1;
  g1[0] = (int)(1u << 16);                                   // wg_mask=0, data_size=1 (2B)
  g1[1] = (int)((width & 0xFFFFu) << 16);                    // tensor_dim0[15:0]
  g1[2] = (int)(((width >> 16) & 0xFFFFu) | ((rows & 0xFFFFu) << 16)); // dim0_hi|dim1_lo
  g1[3] = (int)(((rows >> 16) & 0xFFFFu) | ((width & 0xFFFFu) << 16)); // dim1_hi|tile_dim0
  g1[4] = (int)(rows & 0xFFFFu);                             // tile_dim1 (tile_dim2=0)
  g1[5] = (int)stride;                                       // tensor_dim0_stride[31:0]
  g1[6] = 0;                                                 // stride hi / dim1_stride lo
  g1[7] = 0;
  v4i z4 = {0, 0, 0, 0};
#if defined(__clang_major__) && (__clang_major__ >= 23)
  v8i z8 = {0, 0, 0, 0, 0, 0, 0, 0};
  __builtin_amdgcn_tensor_load_to_lds(g0, g1, z4, z4, z8, 0);
#else
  __builtin_amdgcn_tensor_load_to_lds(g0, g1, z4, z4, 0);
#endif
}

// ---------------------------------------------------------------------------
// Utility kernels
// ---------------------------------------------------------------------------
__global__ void k_f32_to_f16(const float* __restrict__ s, _Float16* __restrict__ d, size_t n) {
  size_t i = (size_t)blockIdx.x * blockDim.x + threadIdx.x;
  size_t stride = (size_t)gridDim.x * blockDim.x;
  for (; i < n; i += stride) d[i] = (_Float16)s[i];
}

__global__ void k_zero_f32(float* __restrict__ p, size_t n) {
  size_t i = (size_t)blockIdx.x * blockDim.x + threadIdx.x;
  size_t stride = (size_t)gridDim.x * blockDim.x;
  for (; i < n; i += stride) p[i] = 0.0f;
}

// out[i] += expert_out[i] * weights[token, p]
__global__ void k_accum_weighted(const float* __restrict__ eo, const float* __restrict__ w,
                                 float* __restrict__ out, int p, size_t n) {
  size_t i = (size_t)blockIdx.x * blockDim.x + threadIdx.x;
  size_t stride = (size_t)gridDim.x * blockDim.x;
  for (; i < n; i += stride) {
    size_t t = i >> 9;                 // / D (=512)
    out[i] += eo[i] * w[t * PP + p];
  }
}

// ---------------------------------------------------------------------------
// Blocked WMMA GEMM:  C[M,N] = A[M,K](f16) @ W[N,K](f16)^T + bias, opt ReLU.
// One wave -> 32x64 C macro-tile (2 A-frags x 4 B-frags -> 8 WMMAs / K-step),
// amortizing fragment loads across 8 wmma issues.  Dual epilogue: writes f16
// (C16) when the consumer is the next f16 stage, else f32 (C32).
// Requires M%32==0, N%64==0, K%32==0.
// ---------------------------------------------------------------------------
__global__ void k_wmma_gemm_big(const _Float16* __restrict__ A, const _Float16* __restrict__ W,
                                const float* __restrict__ bias,
                                float* __restrict__ C32, _Float16* __restrict__ C16,
                                int M, int N, int K, int relu) {
  const int lane = threadIdx.x & 31;
  const int wave = threadIdx.x >> 5;
  const int tile = blockIdx.x * 8 + wave;
  const int ntN  = N >> 6;                       // N/64
  const int ntiles = (M >> 5) * ntN;
  if (tile >= ntiles) return;                    // wave-uniform, EXEC stays all-1s
  const int tm = tile / ntN, tn = tile - tm * ntN;
  const int mBase = tm << 5, nBase = tn << 6;
  const int hi16 = lane >> 4;
  const int l15  = lane & 15;
  const int kbA  = hi16 ? 8 : 0;
  const int kbB  = hi16 ? 16 : 0;

  size_t rowA[2], rowW[4];
#pragma unroll
  for (int mi = 0; mi < 2; ++mi) rowA[mi] = (size_t)(mBase + mi * 16 + l15) * (size_t)K;
#pragma unroll
  for (int ni = 0; ni < 4; ++ni) rowW[ni] = (size_t)(nBase + ni * 16 + l15) * (size_t)K;

  v8f acc[2][4];
#pragma unroll
  for (int mi = 0; mi < 2; ++mi)
#pragma unroll
    for (int ni = 0; ni < 4; ++ni) { v8f z = {}; acc[mi][ni] = z; }

  for (int k0 = 0; k0 < K; k0 += 32) {
    FragU a[2], b[4];
#pragma unroll
    for (int j = 0; j < 8; ++j) {
      const int kj = (j < 4) ? (kbA + 2 * j) : (kbA + 16 + 2 * (j - 4));
#pragma unroll
      for (int mi = 0; mi < 2; ++mi)
        a[mi].u[j] = *(const uint32_t*)(A + rowA[mi] + (size_t)(k0 + kj));
#pragma unroll
      for (int ni = 0; ni < 4; ++ni)
        b[ni].u[j] = *(const uint32_t*)(W + rowW[ni] + (size_t)(k0 + kj));
    }
    if (k0 + 32 < K) {                           // global_prefetch_b8 next K slab
      __builtin_prefetch(A + rowA[0] + (size_t)(k0 + 32), 0, 1);
      __builtin_prefetch(W + rowW[0] + (size_t)(k0 + 32), 0, 1);
    }
#pragma unroll
    for (int mi = 0; mi < 2; ++mi)
#pragma unroll
      for (int ni = 0; ni < 4; ++ni)
        acc[mi][ni] = __builtin_amdgcn_wmma_f32_16x16x32_f16(
            false, a[mi].v, false, b[ni].v, (short)0, acc[mi][ni], false, false);
  }

#pragma unroll
  for (int ni = 0; ni < 4; ++ni) {
    const int col = nBase + ni * 16 + l15;
    const float bv = bias[col];
#pragma unroll
    for (int mi = 0; mi < 2; ++mi) {
#pragma unroll
      for (int v = 0; v < 8; ++v) {
        const int m = mBase + mi * 16 + v + (hi16 ? 8 : 0);
        float val = acc[mi][ni][v] + bv;
        if (relu) val = fmaxf(val, 0.0f);
        if (C16) C16[(size_t)m * N + col] = (_Float16)val;
        else     C32[(size_t)m * N + col] = val;
      }
    }
  }
}

// ---------------------------------------------------------------------------
// Generic 16x16-per-wave WMMA GEMM (used only for N=16 router head), f32 out.
// ---------------------------------------------------------------------------
__global__ void k_wmma_gemm(const _Float16* __restrict__ A, const _Float16* __restrict__ W,
                            const float* __restrict__ bias, float* __restrict__ C,
                            int M, int N, int K, int relu) {
  const int lane = threadIdx.x & 31;
  const int wave = threadIdx.x >> 5;
  const int tile = blockIdx.x * 8 + wave;
  const int ntN  = N >> 4;
  const int ntiles = (M >> 4) * ntN;
  if (tile >= ntiles) return;
  const int tm = tile / ntN, tn = tile - tm * ntN;
  const int mBase = tm << 4, nBase = tn << 4;
  const int hi16 = lane >> 4;
  const int l15  = lane & 15;
  const size_t rowA = (size_t)(mBase + l15) * (size_t)K;
  const size_t rowW = (size_t)(nBase + l15) * (size_t)K;
  const int kbA = hi16 ? 8 : 0;
  const int kbB = hi16 ? 16 : 0;

  v8f acc = {};
  for (int k0 = 0; k0 < K; k0 += 32) {
    FragU a, b;
#pragma unroll
    for (int j = 0; j < 8; ++j) {
      const int kj = (j < 4) ? (kbA + 2 * j) : (kbA + 16 + 2 * (j - 4));
      a.u[j] = *(const uint32_t*)(A + rowA + (size_t)(k0 + kj));
      b.u[j] = *(const uint32_t*)(W + rowW + (size_t)(k0 + kj));
    }
    acc = __builtin_amdgcn_wmma_f32_16x16x32_f16(false, a.v, false, b.v,
                                                 (short)0, acc, false, false);
  }
  const int nCol = nBase + l15;
  const float bv = bias[nCol];
#pragma unroll
  for (int v = 0; v < 8; ++v) {
    const int m = mBase + v + (hi16 ? 8 : 0);
    float val = acc[v] + bv;
    if (relu) val = fmaxf(val, 0.0f);
    C[(size_t)m * N + nCol] = val;
  }
}

// ---------------------------------------------------------------------------
// Flash attention (online softmax), WMMA for QK^T and P*V, TDM for K/V tiles.
// qkv: [M, 3*D] f16 (q | k | v thirds, head h at columns h*DH..).
// One wave per (b, h, 16-query tile).  K/V streamed in 32-key blocks via
// tensor_load_to_lds; expert path (DH=64) double-buffers so TDM overlaps WMMA.
// Output written f16 (feeds the out-proj GEMM directly).
// ---------------------------------------------------------------------------
template <int DH>
__global__ void k_flash_attn(const _Float16* __restrict__ qkv, _Float16* __restrict__ out,
                             int nheads) {
  constexpr bool DB = (DH == 64);                // double-buffer the expert path
  constexpr int NB = DB ? 2 : 1;
  __shared__ alignas(16) _Float16 lQ[16 * DH];
  __shared__ alignas(16) _Float16 lK[NB * 32 * DH];
  __shared__ alignas(16) _Float16 lV[NB * 32 * DH];
  __shared__ alignas(16) _Float16 lP[16 * 32];

  const int wg  = blockIdx.x;
  const int nqt = SS / 16;
  const int qt  = wg % nqt;
  const int h   = (wg / nqt) % nheads;
  const int b   = wg / (nqt * nheads);

  const int lane = threadIdx.x;
  const int hi16 = lane >> 4;
  const int l15  = lane & 15;
  const int kbA  = hi16 ? 8 : 0;
  const int kbB  = hi16 ? 16 : 0;

  const int qoff = h * DH;
  const int koff = DD + h * DH;
  const int voff = 2 * DD + h * DH;
  const int qrow0 = b * SS + qt * 16;
  const int nblk = SS / 32;

  auto issueKV = [&](int blk, int buf) {
    const int krow0 = b * SS + blk * 32;
    tdm_load_2d_f16((uint32_t)(uintptr_t)&lK[buf * 32 * DH],
                    qkv + (size_t)krow0 * TD + koff, DH, 32, TD);
    tdm_load_2d_f16((uint32_t)(uintptr_t)&lV[buf * 32 * DH],
                    qkv + (size_t)krow0 * TD + voff, DH, 32, TD);
  };

  // stage Q tile (16 x DH) into LDS
  for (int w = lane; w < 16 * (DH / 2); w += 32) {
    const int r = w / (DH / 2), c2 = w % (DH / 2);
    *(uint32_t*)&lQ[r * DH + 2 * c2] =
        *(const uint32_t*)(qkv + (size_t)(qrow0 + r) * TD + qoff + 2 * c2);
  }
  if constexpr (DB) issueKV(0, 0);
  __syncthreads();

  float mrow[8], lrow[8];
  v8f O[DH / 16];
#pragma unroll
  for (int v = 0; v < 8; ++v) { mrow[v] = -1e30f; lrow[v] = 0.0f; }
#pragma unroll
  for (int t = 0; t < DH / 16; ++t) { v8f z = {}; O[t] = z; }

  const float scale = 1.0f / sqrtf((float)DH);

  for (int blk = 0; blk < nblk; ++blk) {
    int cur;
    if constexpr (DB) {
      cur = blk & 1;
      if (blk + 1 < nblk) {                      // prefetch next block via TDM
        issueKV(blk + 1, cur ^ 1);
        __builtin_amdgcn_s_wait_tensorcnt(2);    // current block's K,V complete
      } else {
        __builtin_amdgcn_s_wait_tensorcnt(0);
      }
    } else {
      cur = 0;
      issueKV(blk, 0);
      __builtin_amdgcn_s_wait_tensorcnt(0);
    }
    __syncthreads();
    const int kb0 = cur * 32 * DH;

    // ---- S = Q K^T * scale, two 16-key column tiles ----
    v8f s0 = {}, s1 = {};
#pragma unroll
    for (int kk = 0; kk < DH / 32; ++kk) {
      FragU qa, kf0, kf1;
#pragma unroll
      for (int j = 0; j < 8; ++j) {
        const int kj = (j < 4) ? (kbA + 2 * j) : (kbA + 16 + 2 * (j - 4));
        qa.u[j]  = *(const uint32_t*)&lQ[l15 * DH + kk * 32 + kj];
        kf0.u[j] = *(const uint32_t*)&lK[kb0 + (l15)      * DH + kk * 32 + kj];
        kf1.u[j] = *(const uint32_t*)&lK[kb0 + (16 + l15) * DH + kk * 32 + kj];
      }
      s0 = __builtin_amdgcn_wmma_f32_16x16x32_f16(false, qa.v, false, kf0.v,
                                                  (short)0, s0, false, false);
      s1 = __builtin_amdgcn_wmma_f32_16x16x32_f16(false, qa.v, false, kf1.v,
                                                  (short)0, s1, false, false);
    }

    // ---- online softmax update ----
    float alpha[8];
#pragma unroll
    for (int v = 0; v < 8; ++v) {
      const float r0 = s0[v] * scale;
      const float r1 = s1[v] * scale;
      const float bm = fmaxf(rmax16(r0), rmax16(r1));
      const float nm = fmaxf(mrow[v], bm);
      alpha[v] = __expf(mrow[v] - nm);
      mrow[v]  = nm;
      const float p0 = __expf(r0 - nm);
      const float p1 = __expf(r1 - nm);
      s0[v] = p0; s1[v] = p1;
      lrow[v] = lrow[v] * alpha[v] + rsum16(p0) + rsum16(p1);
    }
#pragma unroll
    for (int t = 0; t < DH / 16; ++t)
#pragma unroll
      for (int v = 0; v < 8; ++v) O[t][v] *= alpha[v];

    // ---- stage P (16x32 f16) through LDS, re-read in A-fragment layout ----
#pragma unroll
    for (int v = 0; v < 8; ++v) {
      const int r = v + 8 * hi16;
      lP[r * 32 + l15]      = (_Float16)s0[v];
      lP[r * 32 + 16 + l15] = (_Float16)s1[v];
    }
    __syncthreads();

    FragU pf;
#pragma unroll
    for (int j = 0; j < 8; ++j) {
      const int key = (j < 4) ? (kbA + 2 * j) : (kbA + 16 + 2 * (j - 4));
      pf.u[j] = *(const uint32_t*)&lP[l15 * 32 + key];
    }

    // ---- O += P V ----
#pragma unroll
    for (int t = 0; t < DH / 16; ++t) {
      FragU vf;
      const int dcol = t * 16 + l15;
#pragma unroll
      for (int j = 0; j < 8; ++j) {
        const int key = (j < 4) ? (kbA + 2 * j) : (kbA + 16 + 2 * (j - 4));
        PackU pk;
        pk.h[0] = lV[kb0 + (key)     * DH + dcol];
        pk.h[1] = lV[kb0 + (key + 1) * DH + dcol];
        vf.u[j] = pk.u;
      }
      O[t] = __builtin_amdgcn_wmma_f32_16x16x32_f16(false, pf.v, false, vf.v,
                                                    (short)0, O[t], false, false);
    }
    __syncthreads();
  }

  // ---- normalize and write [M, D] f16 ----
#pragma unroll
  for (int t = 0; t < DH / 16; ++t) {
    const int dcol = h * DH + t * 16 + l15;
#pragma unroll
    for (int v = 0; v < 8; ++v) {
      const int row = qrow0 + v + 8 * hi16;
      out[(size_t)row * DD + dcol] = (_Float16)(O[t][v] / lrow[v]);
    }
  }
}

// ---------------------------------------------------------------------------
// Gating: softmax over P=16, frequency accumulation (GLBL), top-8 scatter
// with jax.lax.top_k tie-breaking (earlier index wins), renormalize.
// ---------------------------------------------------------------------------
__global__ void k_gating(const float* __restrict__ scores, float* __restrict__ freq,
                         float* __restrict__ wout) {
  const int t = blockIdx.x * blockDim.x + threadIdx.x;
  if (t >= MM) return;
  float p[PP];
  float mx = -1e30f;
#pragma unroll
  for (int j = 0; j < PP; ++j) { p[j] = scores[(size_t)t * PP + j]; mx = fmaxf(mx, p[j]); }
  float sum = 0.0f;
#pragma unroll
  for (int j = 0; j < PP; ++j) { p[j] = __expf(p[j] - mx); sum += p[j]; }
  const float inv = 1.0f / sum;
#pragma unroll
  for (int j = 0; j < PP; ++j) p[j] *= inv;
#pragma unroll
  for (int j = 0; j < PP; ++j) atomicAdd(&freq[j], p[j] * (1.0f / (float)MM));

  float kept[PP];
  float ksum = 0.0f;
#pragma unroll
  for (int j = 0; j < PP; ++j) {
    int rank = 0;
#pragma unroll
    for (int i = 0; i < PP; ++i) {
      if (p[i] > p[j]) rank++;
      else if (p[i] == p[j] && i < j) rank++;
    }
    kept[j] = (rank < KK) ? p[j] : 0.0f;
    ksum += kept[j];
  }
  const float rinv = 1.0f / (ksum + 1e-8f);
#pragma unroll
  for (int j = 0; j < PP; ++j) wout[(size_t)t * PP + j] = kept[j] * rinv;
}

__global__ void k_glbl_loss(const float* __restrict__ freq, float* __restrict__ loss) {
  if (threadIdx.x == 0 && blockIdx.x == 0) {
    float s = 0.0f;
#pragma unroll
    for (int j = 0; j < PP; ++j) s += freq[j] * freq[j];
    loss[0] = (float)PP * s;
  }
}

// ---------------------------------------------------------------------------
// Host launcher
// ---------------------------------------------------------------------------
static inline int gs_blocks(size_t n) {
  size_t b = (n + 255) / 256;
  if (b > 16384) b = 16384;
  return (int)b;
}

extern "C" void kernel_launch(void* const* d_in, const int* in_sizes, int n_in,
                              void* d_out, int out_size, void* d_ws, size_t ws_size,
                              hipStream_t stream) {
  (void)in_sizes; (void)n_in; (void)out_size; (void)ws_size;
  const float* x         = (const float*)d_in[0];
  const float* ctx_in_w  = (const float*)d_in[1];
  const float* ctx_in_b  = (const float*)d_in[2];
  const float* ctx_out_w = (const float*)d_in[3];
  const float* ctx_out_b = (const float*)d_in[4];
  const float* r_w1 = (const float*)d_in[5];
  const float* r_b1 = (const float*)d_in[6];
  const float* r_w2 = (const float*)d_in[7];
  const float* r_b2 = (const float*)d_in[8];
  const float* r_w3 = (const float*)d_in[9];
  const float* r_b3 = (const float*)d_in[10];
  const float* p_in_w  = (const float*)d_in[11];
  const float* p_in_b  = (const float*)d_in[12];
  const float* p_out_w = (const float*)d_in[13];
  const float* p_out_b = (const float*)d_in[14];

  float* out      = (float*)d_out;                   // [M*D] output
  float* loss_out = out + (size_t)MM * DD;           // [1]
  float* w_out    = loss_out + 1;                    // [M*P] gating weights

  // ---- workspace layout (bytes) ----
  char* ws = (char*)d_ws;
  constexpr size_t OFF_XH  = 0;                                     // x f16
  constexpr size_t OFF_WH  = OFF_XH + (size_t)MM * DD * 2;          // weights f16
  constexpr size_t WH_N    = (size_t)TD*DD + (size_t)DD*DD + (size_t)RH1_*DD
                           + (size_t)RH2_*RH1_ + (size_t)PP*RH2_
                           + (size_t)PP*TD*DD + (size_t)PP*DD*DD;   // 17,900,544
  constexpr size_t OFF_G16 = OFF_WH  + WH_N * 2;                    // [M,1536] f16
  constexpr size_t OFF_A16 = OFF_G16 + (size_t)MM * TD * 2;         // [M,512]  f16
  constexpr size_t OFF_A32 = OFF_A16 + (size_t)MM * DD * 2;         // [M,512]  f32
  constexpr size_t OFF_H1  = OFF_A32 + (size_t)MM * DD * 4;         // [M,128]  f16
  constexpr size_t OFF_H2  = OFF_H1  + (size_t)MM * RH1_ * 2;       // [M,64]   f16
  constexpr size_t OFF_SC  = OFF_H2  + (size_t)MM * RH2_ * 2;       // [M,16]   f32
  constexpr size_t OFF_FR  = OFF_SC  + (size_t)MM * PP * 4;         // [16]     f32

  _Float16* xh  = (_Float16*)(ws + OFF_XH);
  _Float16* wh  = (_Float16*)(ws + OFF_WH);
  _Float16* g16 = (_Float16*)(ws + OFF_G16);
  _Float16* a16 = (_Float16*)(ws + OFF_A16);
  float*    a32 = (float*)   (ws + OFF_A32);
  _Float16* h1  = (_Float16*)(ws + OFF_H1);
  _Float16* h2  = (_Float16*)(ws + OFF_H2);
  float*    sc  = (float*)   (ws + OFF_SC);
  float*    fr  = (float*)   (ws + OFF_FR);

  // f16 weight sub-offsets (elements)
  const size_t W_CTX_IN  = 0;
  const size_t W_CTX_OUT = W_CTX_IN  + (size_t)TD * DD;
  const size_t W_R1      = W_CTX_OUT + (size_t)DD * DD;
  const size_t W_R2      = W_R1      + (size_t)RH1_ * DD;
  const size_t W_R3      = W_R2      + (size_t)RH2_ * RH1_;
  const size_t W_PIN     = W_R3      + (size_t)PP * RH2_;
  const size_t W_POUT    = W_PIN     + (size_t)PP * TD * DD;

  auto conv = [&](const float* s, _Float16* d, size_t n) {
    k_f32_to_f16<<<gs_blocks(n), 256, 0, stream>>>(s, d, n);
  };
  auto gemmBig = [&](const _Float16* A, const _Float16* W, const float* bias,
                     float* C32, _Float16* C16, int M, int N, int K, int relu) {
    const int tiles = (M / 32) * (N / 64);
    k_wmma_gemm_big<<<(tiles + 7) / 8, 256, 0, stream>>>(A, W, bias, C32, C16,
                                                         M, N, K, relu);
  };

  // ---- precision conversion (x + all weights -> f16, once per call) ----
  conv(x,         xh,             (size_t)MM * DD);
  conv(ctx_in_w,  wh + W_CTX_IN,  (size_t)TD * DD);
  conv(ctx_out_w, wh + W_CTX_OUT, (size_t)DD * DD);
  conv(r_w1,      wh + W_R1,      (size_t)RH1_ * DD);
  conv(r_w2,      wh + W_R2,      (size_t)RH2_ * RH1_);
  conv(r_w3,      wh + W_R3,      (size_t)PP * RH2_);
  conv(p_in_w,    wh + W_PIN,     (size_t)PP * TD * DD);
  conv(p_out_w,   wh + W_POUT,    (size_t)PP * DD * DD);

  // ---- router: context MHA (2 heads, dh=256) ----
  gemmBig(xh, wh + W_CTX_IN, ctx_in_b, nullptr, g16, MM, TD, DD, 0);
  k_flash_attn<256><<<BB * CTXH * (SS / 16), 32, 0, stream>>>(g16, a16, CTXH);
  gemmBig(a16, wh + W_CTX_OUT, ctx_out_b, nullptr, g16, MM, DD, DD, 0);

  // ---- router MLP: D -> 128 -> 64 -> 16 ----
  gemmBig(g16, wh + W_R1, r_b1, nullptr, h1, MM, RH1_, DD, 1);
  gemmBig(h1,  wh + W_R2, r_b2, nullptr, h2, MM, RH2_, RH1_, 1);
  {
    const int tiles = (MM / 16) * (PP / 16);
    k_wmma_gemm<<<(tiles + 7) / 8, 256, 0, stream>>>(h2, wh + W_R3, r_b3, sc,
                                                     MM, PP, RH2_, 0);
  }

  // ---- gating + GLBL loss ----
  k_zero_f32<<<1, 64, 0, stream>>>(fr, PP);
  k_gating<<<(MM + 255) / 256, 256, 0, stream>>>(sc, fr, w_out);
  k_glbl_loss<<<1, 32, 0, stream>>>(fr, loss_out);

  // ---- combine: 16 expert MHAs, weighted accumulate ----
  k_zero_f32<<<gs_blocks((size_t)MM * DD), 256, 0, stream>>>(out, (size_t)MM * DD);
  for (int p = 0; p < PP; ++p) {
    gemmBig(xh, wh + W_PIN + (size_t)p * TD * DD, p_in_b + (size_t)p * TD,
            nullptr, g16, MM, TD, DD, 0);
    k_flash_attn<64><<<BB * HH * (SS / 16), 32, 0, stream>>>(g16, a16, HH);
    gemmBig(a16, wh + W_POUT + (size_t)p * DD * DD, p_out_b + (size_t)p * DD,
            a32, nullptr, MM, DD, DD, 0);
    k_accum_weighted<<<gs_blocks((size_t)MM * DD), 256, 0, stream>>>(
        a32, w_out, out, p, (size_t)MM * DD);
  }
}
